// LSTM_32598801777215
// MI455X (gfx1250) — matrix operand order repaired and checked by reading, not compiled, not run on hardware
//
#include <hip/hip_runtime.h>

typedef __attribute__((ext_vector_type(16))) _Float16 v16h;
typedef __attribute__((ext_vector_type(8)))  float    v8f;

#define BATCH 2048
#define TSTEPS 168
#define HID 128
#define GATES 512
#define KAUG 160   // 128 h + 6 x + 1 bias-one + 25 pad -> 5 k-steps of 32

__device__ __forceinline__ float fsig(float x) {
    // 1 / (1 + e^-x): v_exp_f32 + v_rcp_f32
    return __builtin_amdgcn_rcpf(1.0f + __expf(-x));
}
__device__ __forceinline__ float ftanh_(float x) {
    // tanh(x) = 1 - 2/(1 + e^{2x})
    return 1.0f - 2.0f * __builtin_amdgcn_rcpf(1.0f + __expf(2.0f * x));
}

union AFragU { v16h v; uint4 q[2]; };

__global__ __launch_bounds__(128, 1)
void lstm_persist_wmma(const float* __restrict__ s_inp, const float* __restrict__ flow_x,
                       const float* __restrict__ W_ih, const float* __restrict__ W_hh,
                       const float* __restrict__ b_ih, const float* __restrict__ b_hh,
                       const float* __restrict__ W1,   const float* __restrict__ b1,
                       const float* __restrict__ a1p,  const float* __restrict__ W2,
                       const float* __restrict__ b2,   const float* __restrict__ a2p,
                       float* __restrict__ out)
{
    __shared__ __align__(16) _Float16 Afrag[16][KAUG];   // augmented A rows [h | x | 1 | 0]
    __shared__ float gatesLds[16][GATES];                // f32 gate pre-activations
    __shared__ float hbuf[16][HID];                      // final h (f32) for the MLP head
    __shared__ float y1buf[16][64];

    const int tid  = threadIdx.x;
    const int lane = tid & 31;
    const int wv   = tid >> 5;          // wave 0..3
    const bool hi  = lane >= 16;
    const int ncol = lane & 15;
    const int b0   = blockIdx.x * 16;   // batch tile base

    // ---- Build B fragments (augmented weights, f16) in registers: 8 N-tiles x 5 K-steps.
    // B k order used here (16-bit, 32x16, contiguous, same as A): VGPR j holds K = 2j,2j+1 (lanes 0-15) / K = 16+2j,+1 (lanes 16-31), N = lane&15.
    v16h Bf[8][5];
    #pragma unroll
    for (int n = 0; n < 8; ++n) {
        const int Ng = (wv * 8 + n) * 16 + ncol;   // global gate row 0..511 (i,f,g,o order)
        #pragma unroll
        for (int kk = 0; kk < 5; ++kk) {
            v16h bfv;
            #pragma unroll
            for (int j = 0; j < 8; ++j) {
                const int K0 = kk * 32 + (hi ? 16 : 0) + 2 * j;
                float f0, f1;
                {   // augmented weight row: [W_hh(128) | W_ih(6) | b_ih+b_hh | zeros]
                    int K = K0;
                    f0 = (K < 128) ? W_hh[Ng * 128 + K]
                       : (K < 134) ? W_ih[Ng * 6 + (K - 128)]
                       : (K == 134) ? (b_ih[Ng] + b_hh[Ng]) : 0.0f;
                    K = K0 + 1;
                    f1 = (K < 128) ? W_hh[Ng * 128 + K]
                       : (K < 134) ? W_ih[Ng * 6 + (K - 128)]
                       : (K == 134) ? (b_ih[Ng] + b_hh[Ng]) : 0.0f;
                }
                bfv[2 * j]     = (_Float16)f0;
                bfv[2 * j + 1] = (_Float16)f1;
            }
            Bf[n][kk] = bfv;
        }
    }

    // ---- Init A buffer: h0 = 0, bias-one column, pad zeros, x(t=0)
    for (int i = tid; i < 16 * KAUG; i += 128)
        ((_Float16*)Afrag)[i] = (_Float16)0.0f;
    __syncthreads();
    if (tid < 16) Afrag[tid][134] = (_Float16)1.0f;
    if (tid < 96) {
        const int r = tid / 6, cc = tid % 6;
        const float xv = (cc < 5) ? s_inp[((b0 + r) * TSTEPS + 0) * 5 + cc]
                                  : flow_x[(b0 + r) * TSTEPS + 0];
        Afrag[r][128 + cc] = (_Float16)xv;
    }
    __syncthreads();

    float cst[16];
    #pragma unroll
    for (int r = 0; r < 16; ++r) cst[r] = 0.0f;

    const int mrow  = ncol;            // A-matrix row this lane reads
    const int koffA = hi ? 16 : 0;     // A layout: hi half-wave holds K+16 (contiguous 16 K per half)

    for (int t = 0; t < TSTEPS; ++t) {
        // Prefetch x(t+1) from global; latency hidden under the WMMA phase.
        float xv = 0.0f; int xr = 0, xc = 0;
        if (tid < 96 && t + 1 < TSTEPS) {
            xr = tid / 6; xc = tid % 6;
            xv = (xc < 5) ? s_inp[((b0 + xr) * TSTEPS + (t + 1)) * 5 + xc]
                          : flow_x[(b0 + xr) * TSTEPS + (t + 1)];
        }

        // Load A fragments: two ds_load_b128 per K-step (K=base..base+7, base+8..+15).
        AFragU af[5];
        #pragma unroll
        for (int kk = 0; kk < 5; ++kk) {
            af[kk].q[0] = *(const uint4*)&Afrag[mrow][kk * 32 + koffA];
            af[kk].q[1] = *(const uint4*)&Afrag[mrow][kk * 32 + koffA + 8];
        }

        // WMMA phase: this wave's 8 gate tiles, K reduced over 5 steps of 32.
        #pragma unroll
        for (int n = 0; n < 8; ++n) {
            v8f acc = {0.f, 0.f, 0.f, 0.f, 0.f, 0.f, 0.f, 0.f};
            #pragma unroll
            for (int kk = 0; kk < 5; ++kk)
                acc = __builtin_amdgcn_wmma_f32_16x16x32_f16(
                          false, af[kk].v, false, Bf[n][kk], (short)0, acc, false, false);
            const int gcol = (wv * 8 + n) * 16 + ncol;
            #pragma unroll
            for (int v = 0; v < 8; ++v)                    // C/D layout: M = v + 8*hi, N = lane&15
                gatesLds[v + (hi ? 8 : 0)][gcol] = acc[v];
        }
        __syncthreads();

        // Elementwise LSTM cell: lane owns hidden unit u across all 16 rows.
        const int u = wv * 32 + lane;
        #pragma unroll
        for (int r = 0; r < 16; ++r) {
            const float ig = fsig  (gatesLds[r][u]);
            const float fg = fsig  (gatesLds[r][HID + u]);
            const float gg = ftanh_(gatesLds[r][2 * HID + u]);
            const float og = fsig  (gatesLds[r][3 * HID + u]);
            cst[r] = fg * cst[r] + ig * gg;
            const float hh = og * ftanh_(cst[r]);
            Afrag[r][u] = (_Float16)hh;                    // h feeds next step's A matrix
            if (t == TSTEPS - 1) hbuf[r][u] = hh;          // keep f32 h for the head
        }
        if (tid < 96 && t + 1 < TSTEPS)
            Afrag[xr][128 + xc] = (_Float16)xv;            // next step's x columns
        __syncthreads();
    }

    // ---- MLP head: y1 = PReLU(h @ W1^T + b1), y = PReLU(y1 @ W2^T + b2)
    const float a1v = a1p[0];
    for (int idx = tid; idx < 16 * 64; idx += 128) {
        const int r = idx >> 6, j = idx & 63;
        float acc = b1[j];
        #pragma unroll 8
        for (int uu = 0; uu < 128; ++uu)
            acc += hbuf[r][uu] * W1[j * 128 + uu];
        y1buf[r][j] = (acc >= 0.0f) ? acc : a1v * acc;
    }
    __syncthreads();
    if (tid < 16) {
        const float a2v = a2p[0];
        float acc = b2[0];
        #pragma unroll 8
        for (int j = 0; j < 64; ++j)
            acc += y1buf[tid][j] * W2[j];
        out[b0 + tid] = (acc >= 0.0f) ? acc : a2v * acc;
    }
}

extern "C" void kernel_launch(void* const* d_in, const int* in_sizes, int n_in,
                              void* d_out, int out_size, void* d_ws, size_t ws_size,
                              hipStream_t stream) {
    (void)in_sizes; (void)n_in; (void)out_size; (void)d_ws; (void)ws_size;
    lstm_persist_wmma<<<BATCH / 16, 128, 0, stream>>>(
        (const float*)d_in[0],  // s_inp   [2048,168,5]
        (const float*)d_in[1],  // flow_x  [2048,168,1]
        (const float*)d_in[2],  // W_ih    [512,6]
        (const float*)d_in[3],  // W_hh    [512,128]
        (const float*)d_in[4],  // b_ih    [512]
        (const float*)d_in[5],  // b_hh    [512]
        (const float*)d_in[6],  // W1      [64,128]
        (const float*)d_in[7],  // b1      [64]
        (const float*)d_in[8],  // a1      scalar
        (const float*)d_in[9],  // W2      [1,64]
        (const float*)d_in[10], // b2      scalar
        (const float*)d_in[11], // a2      scalar
        (float*)d_out);         // y       [2048,1]
}
